// GraphLayerLinearSimple_51101520888323
// MI455X (gfx1250) — matrix operand order repaired and checked by reading, not compiled, not run on hardware
//
#include <hip/hip_runtime.h>
#include <math.h>

// GraphLayerLinearSimple — MI455X (gfx1250) implementation.
//
// Strategy:
//  * Factor the pairwise first layer through linearity:
//      hm@Wb1 = h[j]@Wb1_top + h[i]@Wb1_bot  ->  P1[n,j] + P2[n,i]
//    (same for the gate: G1 + G2). This removes the 524k-row K=128 GEMMs.
//  * Stage 1 (tiny, VALU): h = x@W_lin+b; P1,P2,G1,G2; fused unary branch U.
//  * Stage 2 (dominant, WMMA f16): per pair (i,j):
//      out^T += (Wb2^T x tanh(P1[j]+P2[i])^T) * sigmoid-gate, using
//      v_wmma_f32_16x16x32_f16 with the weights as the A operand (resident
//      fragments) and the tanh activations as the B operand (contiguous
//      per-lane loads). sigmoid(z) = 0.5 + 0.5*tanh(z/2), 0.5 folded into
//      Wbg2/bbg2. Biases folded via a separate sum-of-gates accumulator.

typedef __attribute__((ext_vector_type(16))) _Float16 v16h;
typedef __attribute__((ext_vector_type(8)))  float    v8f;

#define C_DIM 64
#define KK    64
#define NN    128
#define ROWS  (NN * KK)   /* 8192 */

__device__ __forceinline__ float tanh_fast(float x) {
#if __has_builtin(__builtin_amdgcn_tanhf)
  return __builtin_amdgcn_tanhf(x);
#elif __has_builtin(__builtin_amdgcn_tanh_f32)
  return __builtin_amdgcn_tanh_f32(x);
#else
  return tanhf(x);
#endif
}

// ---------------------------------------------------------------------------
// Stage 1: h = x@W_lin + b_lin, then
//   P1 = h@Wb1[0:64]  + bb1   (j-side, bias folded)
//   P2 = h@Wb1[64:128]        (i-side)
//   G1 = h@Wbg1[0:64] + bbg1
//   G2 = h@Wbg1[64:128]
//   U  = (tanh(h@Wu1+bu1)@Wu2+bu2) * sigmoid(tanh(h@Wug1+bug1)@Wug2+bug2)
// One block per row (64 threads = one output channel each).
// ---------------------------------------------------------------------------
__global__ __launch_bounds__(64)
void stage1_kernel(const float* __restrict__ x,
                   const float* __restrict__ Wlin, const float* __restrict__ blin,
                   const float* __restrict__ Wu1,  const float* __restrict__ bu1,
                   const float* __restrict__ Wu2,  const float* __restrict__ bu2,
                   const float* __restrict__ Wug1, const float* __restrict__ bug1,
                   const float* __restrict__ Wug2, const float* __restrict__ bug2,
                   const float* __restrict__ Wb1,  const float* __restrict__ bb1,
                   const float* __restrict__ Wbg1, const float* __restrict__ bbg1,
                   float* __restrict__ P1, float* __restrict__ P2,
                   float* __restrict__ G1, float* __restrict__ G2,
                   float* __restrict__ U)
{
  __shared__ float xs[C_DIM], hs[C_DIM], a1s[C_DIM], ags[C_DIM];
  const int row = blockIdx.x;
  const int c   = threadIdx.x;

  xs[c] = x[row * C_DIM + c];
  __syncthreads();

  float h = blin[c];
#pragma unroll 8
  for (int k = 0; k < C_DIM; ++k) h += xs[k] * Wlin[k * C_DIM + c];
  hs[c] = h;
  __syncthreads();

  float p1 = bb1[c], p2 = 0.f, g1 = bbg1[c], g2 = 0.f;
  float u1 = bu1[c], w1 = bug1[c];
#pragma unroll 4
  for (int k = 0; k < C_DIM; ++k) {
    const float hk = hs[k];
    p1 += hk * Wb1 [k * C_DIM + c];
    p2 += hk * Wb1 [(C_DIM + k) * C_DIM + c];
    g1 += hk * Wbg1[k * C_DIM + c];
    g2 += hk * Wbg1[(C_DIM + k) * C_DIM + c];
    u1 += hk * Wu1 [k * C_DIM + c];
    w1 += hk * Wug1[k * C_DIM + c];
  }
  const int o = row * C_DIM + c;
  P1[o] = p1; P2[o] = p2; G1[o] = g1; G2[o] = g2;
  a1s[c] = tanh_fast(u1);
  ags[c] = tanh_fast(w1);
  __syncthreads();

  float um = bu2[c], ug = bug2[c];
#pragma unroll 8
  for (int k = 0; k < C_DIM; ++k) {
    um += a1s[k] * Wu2 [k * C_DIM + c];
    ug += ags[k] * Wug2[k * C_DIM + c];
  }
  U[o] = um * (0.5f + 0.5f * tanh_fast(0.5f * ug));   // sigmoid via tanh
}

// ---------------------------------------------------------------------------
// Stage 2: pairwise WMMA kernel.
// Grid: 512 blocks = (n, j-tile of 16). Block = 128 threads = 4 waves;
// wave w owns i in [16w, 16w+16). Computes out^T = Wb2^T x t1^T per i,
// gates element-wise, accumulates, then LDS cross-wave reduction.
//
// WMMA 16x16x32 f16 operand layouts as used here (A/B k order chosen by this code):
//  A (16x32, weights): lane l holds row M=l&15; elem e -> K = 16*(l>>4)+e   (contiguous, matches B)
//  B (32x16, activ.) : lane l holds col N=l&15; elem e -> K = 16*(l>>4)+e   (contiguous; not the ISA table order)
//  C/D (16x16 f32)   : lane l col N=l&15; vgpr r -> row M = r + 8*(l>>4)
// ---------------------------------------------------------------------------
__global__ __launch_bounds__(128, 1)
void pair_wmma_kernel(const float* __restrict__ P1, const float* __restrict__ P2,
                      const float* __restrict__ G1, const float* __restrict__ G2,
                      const float* __restrict__ U,
                      const float* __restrict__ Wb2,  const float* __restrict__ bb2,
                      const float* __restrict__ Wbg2, const float* __restrict__ bbg2,
                      float* __restrict__ out)
{
  const int n    = blockIdx.x >> 2;
  const int j0   = (blockIdx.x & 3) << 4;
  const int tid  = threadIdx.x;
  const int wave = tid >> 5;
  const int lane = tid & 31;
  const int hi   = lane >> 4;
  const int jl   = lane & 15;

  // Resident weight A-fragments: Aw = Wb2^T tiles, Ag = (0.5*Wbg2)^T tiles.
  v16h Aw[4][2], Ag[4][2];
#pragma unroll
  for (int m = 0; m < 4; ++m)
#pragma unroll
    for (int s = 0; s < 2; ++s)
#pragma unroll
      for (int e = 0; e < 16; ++e) {
        const int cin  = s * 32 + 16 * hi + e;
        const int cout = m * 16 + jl;
        Aw[m][s][e] = (_Float16)(Wb2 [cin * 64 + cout]);
        Ag[m][s][e] = (_Float16)(0.5f * Wbg2[cin * 64 + cout]);
      }

  // Gate bias per D element (already halved for the sigmoid-as-tanh trick).
  float bgb[4][8];
#pragma unroll
  for (int m = 0; m < 4; ++m)
#pragma unroll
    for (int r = 0; r < 8; ++r)
      bgb[m][r] = 0.5f * bbg2[m * 16 + 8 * hi + r];

  // Resident j-side pre-activations for row j = j0 + jl (B-layout c ranges).
  const int rowj = (n * 64 + j0 + jl) * 64;
  float p1r[2][16], g1r[2][16];
#pragma unroll
  for (int s = 0; s < 2; ++s) {
    const int cb = s * 32 + hi * 16;
#pragma unroll
    for (int e = 0; e < 16; ++e) {
      p1r[s][e] = P1[rowj + cb + e];
      g1r[s][e] = G1[rowj + cb + e];
    }
  }

  v8f accm[4] = {};   // sum_i D_i * g_i
  v8f accg[4] = {};   // sum_i g_i   (for deferred bb2 bias)

  const int i0 = wave * 16;
  for (int i = i0; i < i0 + 16; ++i) {
    const int rowi = (n * 64 + i) * 64;

    // Build B fragments: t1 = tanh(P1[j]+P2[i]),  t2 = tanh(G1[j]+G2[i]).
    v16h B1[2], B2[2];
#pragma unroll
    for (int s = 0; s < 2; ++s) {
      const float4* p2v = (const float4*)(P2 + rowi + s * 32 + hi * 16);
      const float4* g2v = (const float4*)(G2 + rowi + s * 32 + hi * 16);
#pragma unroll
      for (int q = 0; q < 4; ++q) {
        const float4 a = p2v[q];
        const float4 b = g2v[q];
        B1[s][q * 4 + 0] = (_Float16)tanh_fast(p1r[s][q * 4 + 0] + a.x);
        B1[s][q * 4 + 1] = (_Float16)tanh_fast(p1r[s][q * 4 + 1] + a.y);
        B1[s][q * 4 + 2] = (_Float16)tanh_fast(p1r[s][q * 4 + 2] + a.z);
        B1[s][q * 4 + 3] = (_Float16)tanh_fast(p1r[s][q * 4 + 3] + a.w);
        B2[s][q * 4 + 0] = (_Float16)tanh_fast(g1r[s][q * 4 + 0] + b.x);
        B2[s][q * 4 + 1] = (_Float16)tanh_fast(g1r[s][q * 4 + 1] + b.y);
        B2[s][q * 4 + 2] = (_Float16)tanh_fast(g1r[s][q * 4 + 2] + b.z);
        B2[s][q * 4 + 3] = (_Float16)tanh_fast(g1r[s][q * 4 + 3] + b.w);
      }
    }

    // 16 WMMAs per i: 4 M-tiles x 2 k-steps x {message, gate}.
#pragma unroll
    for (int m = 0; m < 4; ++m) {
      v8f d = {};
      d = __builtin_amdgcn_wmma_f32_16x16x32_f16(false, Aw[m][0], false, B1[0],
                                                 (short)0, d, false, false);
      d = __builtin_amdgcn_wmma_f32_16x16x32_f16(false, Aw[m][1], false, B1[1],
                                                 (short)0, d, false, false);
      v8f dg = {};
      dg = __builtin_amdgcn_wmma_f32_16x16x32_f16(false, Ag[m][0], false, B2[0],
                                                  (short)0, dg, false, false);
      dg = __builtin_amdgcn_wmma_f32_16x16x32_f16(false, Ag[m][1], false, B2[1],
                                                  (short)0, dg, false, false);
#pragma unroll
      for (int r = 0; r < 8; ++r) {
        const float g = 0.5f + 0.5f * tanh_fast(dg[r] + bgb[m][r]); // sigmoid
        accm[m][r] += d[r] * g;
        accg[m][r] += g;
      }
    }
  }

  // Cross-wave reduction over i-partials, add unary branch + deferred bias.
  __shared__ float lm[4][16][64];
  __shared__ float lg[4][16][64];
#pragma unroll
  for (int m = 0; m < 4; ++m)
#pragma unroll
    for (int r = 0; r < 8; ++r) {
      const int cout = m * 16 + 8 * hi + r;
      lm[wave][jl][cout] = accm[m][r];
      lg[wave][jl][cout] = accg[m][r];
    }
  __syncthreads();

  const float invk = 1.0f / 63.0f;   // 1/(K-1)
  for (int idx = tid; idx < 16 * 64; idx += 128) {
    const int j = idx >> 6;
    const int c = idx & 63;
    const float am = lm[0][j][c] + lm[1][j][c] + lm[2][j][c] + lm[3][j][c];
    const float ag = lg[0][j][c] + lg[1][j][c] + lg[2][j][c] + lg[3][j][c];
    const int o = (n * 64 + j0 + j) * 64 + c;
    out[o] = U[o] + (am + bb2[c] * ag) * invk;   // sum_i (D+bb2)*g = sumDg + bb2*sumg
  }
}

// ---------------------------------------------------------------------------
extern "C" void kernel_launch(void* const* d_in, const int* in_sizes, int n_in,
                              void* d_out, int out_size, void* d_ws, size_t ws_size,
                              hipStream_t stream)
{
  (void)in_sizes; (void)n_in; (void)out_size; (void)ws_size;
  const float* x    = (const float*)d_in[0];
  const float* Wlin = (const float*)d_in[1];
  const float* blin = (const float*)d_in[2];
  const float* Wu1  = (const float*)d_in[3];
  const float* bu1  = (const float*)d_in[4];
  const float* Wu2  = (const float*)d_in[5];
  const float* bu2  = (const float*)d_in[6];
  const float* Wug1 = (const float*)d_in[7];
  const float* bug1 = (const float*)d_in[8];
  const float* Wug2 = (const float*)d_in[9];
  const float* bug2 = (const float*)d_in[10];
  const float* Wb1  = (const float*)d_in[11];
  const float* bb1  = (const float*)d_in[12];
  const float* Wb2  = (const float*)d_in[13];
  const float* bb2  = (const float*)d_in[14];
  const float* Wbg1 = (const float*)d_in[15];
  const float* bbg1 = (const float*)d_in[16];
  const float* Wbg2 = (const float*)d_in[17];
  const float* bbg2 = (const float*)d_in[18];
  // d_in[19] = t, ignored (IgnoreLinear).

  float* ws = (float*)d_ws;
  const size_t S = (size_t)ROWS * C_DIM;          // 524288 floats = 2 MB
  float* P1 = ws;          float* P2 = ws + S;
  float* G1 = ws + 2 * S;  float* G2 = ws + 3 * S;
  float* Uu = ws + 4 * S;                         // total 10 MB scratch

  stage1_kernel<<<ROWS, 64, 0, stream>>>(x, Wlin, blin, Wu1, bu1, Wu2, bu2,
                                         Wug1, bug1, Wug2, bug2,
                                         Wb1, bb1, Wbg1, bbg1,
                                         P1, P2, G1, G2, Uu);

  pair_wmma_kernel<<<NN * 4, 128, 0, stream>>>(P1, P2, G1, G2, Uu,
                                               Wb2, bb2, Wbg2, bbg2,
                                               (float*)d_out);
}
